// PositionProjection_44040594653319
// MI455X (gfx1250) — compile-verified, matrix operand order checked by reading, not run on hardware
//
#include <hip/hip_runtime.h>
#include <hip/hip_bf16.h>

// ---------------------------------------------------------------------------
// PositionProjection on MI455X (gfx1250, wave32, WMMA)
//   t   = pe @ proj            [8192,768]x[768,128]  (bf16 WMMA, fp32 acc)
//   d2  = sq[m] + sq[n] - 2 * (t @ t^T)              (bf16 WMMA, fp32 acc)
// Output 8192^2 fp32 = 256 MB -> HBM-store bound (~11.5us @ 23.3 TB/s).
// t (2 MB bf16) is L2-resident; Gram compute via v_wmma_f32_16x16x32_bf16
// stays well under the memory floor. Output uses non-temporal stores.
// ---------------------------------------------------------------------------

typedef __attribute__((ext_vector_type(16))) __bf16 v16bf;
typedef __attribute__((ext_vector_type(8)))  __bf16 v8bf;
typedef __attribute__((ext_vector_type(8)))  float  v8f;
typedef __attribute__((ext_vector_type(4)))  float  v4f;

#define N_PTS 8192
#define D_IN  768
#define RANK  128

static __device__ __forceinline__ v16bf cat16(v8bf lo, v8bf hi) {
  v16bf r;
#pragma unroll
  for (int i = 0; i < 8; ++i) { r[i] = lo[i]; r[i + 8] = hi[i]; }
  return r;
}

// Load one 16-lane-half fragment slice: per ISA 7.12.2, a lane's 16 bf16
// elements for a 16x32 (MxK) A-tile are K = {kbase..kbase+7, 16+kbase..16+kbase+7}
// with kbase = (lane/16)*8 -> two contiguous 16B loads from a row-major row.
static __device__ __forceinline__ v16bf ldfrag(const __bf16* p) {
  v8bf lo = *(const v8bf*)(p);
  v8bf hi = *(const v8bf*)(p + 16);
  return cat16(lo, hi);
}

static __device__ __forceinline__ v8f wmma_bf16(v16bf a, v16bf b, v8f c) {
  return __builtin_amdgcn_wmma_f32_16x16x32_bf16(
      /*neg_a=*/false, a, /*neg_b=*/false, b,
      /*c_mod=*/(short)0, c, /*reuse_a=*/false, /*reuse_b=*/false);
}

// ---------------------------------------------------------------------------
// Kernel 0: projT[r][d] = (bf16) proj[d][r]   (128 x 768, makes B loads linear)
// ---------------------------------------------------------------------------
__global__ void k_transpose_proj(const float* __restrict__ proj,
                                 __bf16* __restrict__ projT) {
  int idx = blockIdx.x * 256 + threadIdx.x;     // 0 .. 128*768-1
  int r = idx / D_IN;
  int d = idx - r * D_IN;
  projT[idx] = (__bf16)proj[(size_t)d * RANK + r];
}

// ---------------------------------------------------------------------------
// Kernel 1: t = pe @ proj  -> bf16 [8192 x 128]
// One wave per 16x16 output tile; 8 waves/block cover all 8 RANK tiles.
// ---------------------------------------------------------------------------
__global__ void k_proj_gemm(const float* __restrict__ pe,
                            const __bf16* __restrict__ projT,
                            __bf16* __restrict__ t) {
  const int lane  = threadIdx.x & 31;
  const int wave  = threadIdx.x >> 5;           // 0..7 -> n-tile
  const int m0    = blockIdx.x * 16;
  const int n0    = wave * 16;
  const int half  = lane >> 4;                  // lane group
  const int l15   = lane & 15;
  const int kbase = half * 8;

  const float*  aptr = pe    + (size_t)(m0 + l15) * D_IN + kbase;
  const __bf16* bptr = projT + (size_t)(n0 + l15) * D_IN + kbase;

  v8f acc = {};
#pragma unroll 4
  for (int k = 0; k < D_IN; k += 32) {
    // A fragment: 16 fp32 -> bf16 (elements 0..7 = K kbase.., 8..15 = K 16+kbase..)
    v4f a0 = *(const v4f*)(aptr + k);
    v4f a1 = *(const v4f*)(aptr + k + 4);
    v4f a2 = *(const v4f*)(aptr + k + 16);
    v4f a3 = *(const v4f*)(aptr + k + 20);
    v16bf a;
#pragma unroll
    for (int i = 0; i < 4; ++i) {
      a[i]      = (__bf16)a0[i];
      a[i + 4]  = (__bf16)a1[i];
      a[i + 8]  = (__bf16)a2[i];
      a[i + 12] = (__bf16)a3[i];
    }
    v16bf b = ldfrag(bptr + k);                 // B[k][n] = projT[n][k]
    acc = wmma_bf16(a, b, acc);
  }

  // D layout: lane holds (m = m0 + half*8 + r, n = n0 + l15), r = 0..7
  const int n = n0 + l15;
#pragma unroll
  for (int r = 0; r < 8; ++r) {
    const int m = m0 + half * 8 + r;
    t[(size_t)m * RANK + n] = (__bf16)acc[r];
  }
}

// ---------------------------------------------------------------------------
// Kernel 2: sq[row] = sum_k t[row][k]^2
// ---------------------------------------------------------------------------
__global__ void k_row_sq(const __bf16* __restrict__ t, float* __restrict__ sq) {
  const int row = blockIdx.x * 256 + threadIdx.x;     // exact: 8192 threads
  const __bf16* p = t + (size_t)row * RANK;
  float s = 0.f;
#pragma unroll
  for (int k = 0; k < RANK; k += 8) {
    v8bf v = *(const v8bf*)(p + k);
#pragma unroll
    for (int i = 0; i < 8; ++i) { float f = (float)v[i]; s += f * f; }
  }
  sq[row] = s;
}

// ---------------------------------------------------------------------------
// Kernel 3: out[m][n] = sq[m] + sq[n] - 2 * (t @ t^T)[m][n]
// Block = 256 threads = 8 waves arranged 4x2; block tile 128x64.
// Each wave computes a 32x32 region = 2x2 WMMA tiles (frag reuse).
// ---------------------------------------------------------------------------
__global__ void k_gram_dist(const __bf16* __restrict__ t,
                            const float* __restrict__ sq,
                            float* __restrict__ out) {
  const int lane  = threadIdx.x & 31;
  const int wave  = threadIdx.x >> 5;
  const int wrow  = wave >> 1;                  // 0..3
  const int wcol  = wave & 1;                   // 0..1
  const int bm    = blockIdx.x * 128 + wrow * 32;
  const int bn    = blockIdx.y * 64  + wcol * 32;
  const int half  = lane >> 4;
  const int l15   = lane & 15;
  const int kbase = half * 8;

  const __bf16* aptr0 = t + (size_t)(bm + l15) * RANK + kbase;
  const __bf16* aptr1 = aptr0 + (size_t)16 * RANK;
  const __bf16* bptr0 = t + (size_t)(bn + l15) * RANK + kbase;  // B[k][n]=t[n][k]
  const __bf16* bptr1 = bptr0 + (size_t)16 * RANK;

  v8f acc00 = {}, acc01 = {}, acc10 = {}, acc11 = {};
#pragma unroll
  for (int k = 0; k < RANK; k += 32) {
    v16bf a0 = ldfrag(aptr0 + k);
    v16bf a1 = ldfrag(aptr1 + k);
    v16bf b0 = ldfrag(bptr0 + k);
    v16bf b1 = ldfrag(bptr1 + k);
    acc00 = wmma_bf16(a0, b0, acc00);
    acc01 = wmma_bf16(a0, b1, acc01);
    acc10 = wmma_bf16(a1, b0, acc10);
    acc11 = wmma_bf16(a1, b1, acc11);
  }

  // Epilogue: d2 = sq[m] + sq[n] - 2*acc ; NT stores (don't pollute L2).
  const int n0 = bn + l15;
  const int n1 = n0 + 16;
  const float sqn0 = sq[n0];
  const float sqn1 = sq[n1];
#pragma unroll
  for (int r = 0; r < 8; ++r) {
    const int m0 = bm + half * 8 + r;
    const int m1 = m0 + 16;
    const float sqm0 = sq[m0];
    const float sqm1 = sq[m1];
    __builtin_nontemporal_store(sqm0 + sqn0 - 2.0f * acc00[r],
                                out + (size_t)m0 * N_PTS + n0);
    __builtin_nontemporal_store(sqm0 + sqn1 - 2.0f * acc01[r],
                                out + (size_t)m0 * N_PTS + n1);
    __builtin_nontemporal_store(sqm1 + sqn0 - 2.0f * acc10[r],
                                out + (size_t)m1 * N_PTS + n0);
    __builtin_nontemporal_store(sqm1 + sqn1 - 2.0f * acc11[r],
                                out + (size_t)m1 * N_PTS + n1);
  }
}

// ---------------------------------------------------------------------------
extern "C" void kernel_launch(void* const* d_in, const int* in_sizes, int n_in,
                              void* d_out, int out_size, void* d_ws, size_t ws_size,
                              hipStream_t stream) {
  const float* pe   = (const float*)d_in[0];   // [8192, 768]
  const float* proj = (const float*)d_in[1];   // [768, 128]
  float* out        = (float*)d_out;           // [8192, 8192]

  // Workspace layout (~2.3 MB total):
  __bf16* t     = (__bf16*)d_ws;               // 8192*128 bf16 = 2 MB
  __bf16* projT = t + (size_t)N_PTS * RANK;    // 128*768 bf16  = 192 KB
  float*  sq    = (float*)(projT + (size_t)RANK * D_IN); // 8192 f32 = 32 KB

  k_transpose_proj<<<(RANK * D_IN) / 256, 256, 0, stream>>>(proj, projT);
  k_proj_gemm<<<N_PTS / 16, 256, 0, stream>>>(pe, projT, t);
  k_row_sq<<<N_PTS / 256, 256, 0, stream>>>(t, sq);
  dim3 grid(N_PTS / 128, N_PTS / 64);
  k_gram_dist<<<grid, 256, 0, stream>>>(t, sq, out);
}
